// EfficientSelfAttention3D_86234353369889
// MI455X (gfx1250) — matrix operand order repaired and checked by reading, not compiled, not run on hardware
//
#include <hip/hip_runtime.h>

typedef __attribute__((ext_vector_type(16))) __bf16 v16bf;
typedef __attribute__((ext_vector_type(8)))  float  v8f;

#define WMMA_BF16(A, B, C) \
  __builtin_amdgcn_wmma_f32_16x16x32_bf16(false, (A), false, (B), (short)0, (C), false, false)

// Issue one per-lane async global->LDS copy of 4 bytes (CDNA5, ASYNCcnt-tracked).
__device__ __forceinline__ void async_copy_b32(const float* gptr, float* lds_generic) {
  const unsigned lds_off = (unsigned)(unsigned long long)lds_generic; // flat->LDS: addr[31:0]
  const unsigned long long gaddr = (unsigned long long)gptr;
  asm volatile("global_load_async_to_lds_b32 %0, %1, off"
               :: "v"(lds_off), "v"(gaddr) : "memory");
}
__device__ __forceinline__ void wait_async_zero() {
  asm volatile("s_wait_asynccnt 0" ::: "memory");
}

// ---------------------------------------------------------------------------
// Kernel 1: kv = conv3d(x, w_kv, kernel=stride=8) as GEMM [128 x 256 x 32768].
// One workgroup (8 waves) per 16-position tile; all 8 filter tiles share the
// patch block staged into LDS via async global->LDS copies. x is read once
// from HBM; w_kv re-reads are L2-resident (16.8 MB << 192 MB L2).
// Writes k_seq [b][h][c][j] bf16 (channel-major => B operand of QK^T)
// and    v_seq [b][h][j][c] bf16 (row-major    => B operand of attn*V).
// ---------------------------------------------------------------------------
__global__ __launch_bounds__(256)
void kv_conv_kernel(const float* __restrict__ x,
                    const float* __restrict__ w_kv,
                    __bf16* __restrict__ k_seq,
                    __bf16* __restrict__ v_seq) {
  __shared__ float xs[16][32];            // staged patch block: [j_local][k_local]

  const int t    = threadIdx.x;
  const int lane = t & 31;
  const int mt   = t >> 5;                // wave id == filter tile (8 waves)
  const int b    = blockIdx.x >> 4;       // 32 blocks = 2 batches * 16 pos-tiles
  const int nt   = blockIdx.x & 15;
  const int l16  = lane & 15;
  const bool hi  = lane >= 16;
  const int kbA  = hi ? 8 : 0;            // unused; A and B fragments both use kbB
  const int kbB  = hi ? 16 : 0;           // B-fragment K base

  const int o = mt * 16 + l16;            // filter row for A loads
  const long xb = (long)b * 64 * 131072;

  // Two staging elements per thread: (j0s, kk) and (j0s+8, kk).
  const int kk  = t & 31;
  const int j0s = t >> 5;                 // 0..7
  long sb[2];
#pragma unroll
  for (int r = 0; r < 2; ++r) {
    const int j = nt * 16 + j0s + r * 8;
    const int jd = j >> 6, jh = (j >> 3) & 7, jw = j & 7;
    sb[r] = (long)jd * 32768 + (long)jh * 512 + (long)jw * 8;
  }

  v8f acc = {};
  for (int k0 = 0; k0 < 32768; k0 += 32) {
    __syncthreads();                      // previous block's LDS reads done
    // ---- async-stage patch block [32 K][16 j] into LDS ----
    {
      const int k = k0 + kk;              // k = ((c*8+dz)*8+dy)*8+dx
      const int c  = k >> 9;
      const int dz = (k >> 6) & 7;
      const int dy = (k >> 3) & 7;
      const int dx = k & 7;
      const long off = (long)c * 131072 + dz * 4096 + dy * 64 + dx;
#pragma unroll
      for (int r = 0; r < 2; ++r)
        async_copy_b32(x + xb + sb[r] + off, &xs[j0s + r * 8][kk]);
    }
    wait_async_zero();
    __syncthreads();                      // staged data visible to all waves

    // ---- fragments + WMMA ----
    v16bf a, bm;
    __builtin_prefetch(w_kv + (long)o * 32768 + k0 + 64, 0, 1);
#pragma unroll
    for (int e = 0; e < 16; ++e) {
      const int ka = k0 + kbB + e;
      a[e]  = (__bf16)w_kv[(long)o * 32768 + ka];
      bm[e] = (__bf16)xs[l16][kbB + e];
    }
    acc = WMMA_BF16(a, bm, acc);
  }

  const int j = nt * 16 + l16;            // output position for this lane's column
#pragma unroll
  for (int r = 0; r < 8; ++r) {
    const int oo = mt * 16 + r + (hi ? 8 : 0);
    const __bf16 val = (__bf16)acc[r];
    if (oo < 64) {                        // K half
      const int h = oo >> 4, c = oo & 15;
      k_seq[(((long)b * 4 + h) * 16 + c) * 256 + j] = val;
    } else {                              // V half
      const int oz = oo - 64;
      const int h = oz >> 4, c = oz & 15;
      v_seq[(((long)b * 4 + h) * 256 + j) * 16 + c] = val;
    }
  }
}

// ---------------------------------------------------------------------------
// Kernel 2: fused q-proj + attention + out-proj.  One wave per 16-position tile.
//   q    = w_q @ x_tile            (WMMA, K=64)
//   sim  = q @ k^T * hd^-0.5       (WMMA per head, K=16 zero-padded to 32)
//   attn = softmax(sim) over 256   (LDS)
//   out  = attn @ v                (WMMA per head, K=256)
//   y    = w_out @ out_all         (WMMA, K=64) -> d_out f32
// ---------------------------------------------------------------------------
__global__ __launch_bounds__(32)
void attn_fused_kernel(const float* __restrict__ x,
                       const float* __restrict__ w_q,
                       const float* __restrict__ w_out,
                       const __bf16* __restrict__ k_seq,
                       const __bf16* __restrict__ v_seq,
                       float* __restrict__ out) {
  __shared__ float sim[16][256];          // 16 KB: scores for one head
  __shared__ float qlds[4][16][16];       // 4 KB : q[pos][c] per head
  __shared__ float outb[64][16];          // 4 KB : out_all[c_all][pos]

  const int lane = threadIdx.x;
  const int l16 = lane & 15;
  const bool hi = lane >= 16;
  const int kbA = hi ? 8 : 0;
  const int kbB = hi ? 16 : 0;

  const long N = 131072;
  const int b  = blockIdx.x >> 13;        // 8192 tiles per batch
  const int nt = blockIdx.x & 8191;
  const long n0 = (long)nt * 16;
  const long n  = n0 + l16;

  // ---- q projection: per-head q tiles into LDS ----
  v16bf bx[2];
#pragma unroll
  for (int kt = 0; kt < 2; ++kt)
#pragma unroll
    for (int e = 0; e < 16; ++e) {
      const int c = kt * 32 + kbB + e;
      bx[kt][e] = (__bf16)x[((long)b * 64 + c) * N + n];
    }
#pragma unroll
  for (int mt = 0; mt < 4; ++mt) {
    v8f acc = {};
#pragma unroll
    for (int kt = 0; kt < 2; ++kt) {
      v16bf a;
#pragma unroll
      for (int e = 0; e < 16; ++e) {
        const int c = kt * 32 + kbB + e;
        a[e] = (__bf16)w_q[(mt * 16 + l16) * 64 + c];
      }
      acc = WMMA_BF16(a, bx[kt], acc);
    }
#pragma unroll
    for (int r = 0; r < 8; ++r)
      qlds[mt][l16][r + (hi ? 8 : 0)] = acc[r];   // [pos][c]
  }
  __syncthreads();

  // ---- attention per head ----
  for (int h = 0; h < 4; ++h) {
    v16bf aq;                             // A = q (M=pos, K=c; c 16..31 zero)
#pragma unroll
    for (int e = 0; e < 16; ++e)
      aq[e] = hi ? (__bf16)0.0f : (__bf16)qlds[h][l16][e];

    for (int jt = 0; jt < 16; ++jt) {     // sim = q @ k^T
      v16bf bk;
#pragma unroll
      for (int e = 0; e < 16; ++e)
        bk[e] = hi ? (__bf16)0.0f
                   : k_seq[(((long)b * 4 + h) * 16 + e) * 256 + jt * 16 + l16];
      v8f s = {};
      s = WMMA_BF16(aq, bk, s);
#pragma unroll
      for (int r = 0; r < 8; ++r)
        sim[r + (hi ? 8 : 0)][jt * 16 + l16] = s[r] * 0.25f;  // hd^-0.5
    }
    __syncthreads();

    if (lane < 16) {                      // row softmax over 256 kv positions
      float m = -1e30f;
      for (int jj = 0; jj < 256; ++jj) m = fmaxf(m, sim[lane][jj]);
      float ssum = 0.0f;
      for (int jj = 0; jj < 256; ++jj) {
        const float e = __expf(sim[lane][jj] - m);
        sim[lane][jj] = e;
        ssum += e;
      }
      const float inv = 1.0f / ssum;
      for (int jj = 0; jj < 256; ++jj) sim[lane][jj] *= inv;
    }
    __syncthreads();

    v8f oacc = {};                        // out = attn @ v (K=256)
    for (int kt = 0; kt < 8; ++kt) {
      v16bf aa, bv;
#pragma unroll
      for (int e = 0; e < 16; ++e) {
        const int jA = kt * 32 + kbB + e;
        aa[e] = (__bf16)sim[l16][jA];
        const int jB = kt * 32 + kbB + e;
        bv[e] = v_seq[(((long)b * 4 + h) * 256 + jB) * 16 + l16];
      }
      oacc = WMMA_BF16(aa, bv, oacc);
    }
#pragma unroll
    for (int r = 0; r < 8; ++r)
      outb[h * 16 + l16][r + (hi ? 8 : 0)] = oacc[r];  // [c_all][pos]
    __syncthreads();
  }

  // ---- output projection: y = w_out @ out_all ----
#pragma unroll
  for (int mt = 0; mt < 4; ++mt) {
    v8f acc = {};
#pragma unroll
    for (int kt = 0; kt < 2; ++kt) {
      v16bf a, bb;
#pragma unroll
      for (int e = 0; e < 16; ++e) {
        const int cA = kt * 32 + kbB + e;
        a[e] = (__bf16)w_out[(mt * 16 + l16) * 64 + cA];
        const int cB = kt * 32 + kbB + e;
        bb[e] = (__bf16)outb[cB][l16];
      }
      acc = WMMA_BF16(a, bb, acc);
    }
#pragma unroll
    for (int r = 0; r < 8; ++r)
      out[((long)b * 64 + mt * 16 + r + (hi ? 8 : 0)) * N + n] = acc[r];
  }
}

// ---------------------------------------------------------------------------
extern "C" void kernel_launch(void* const* d_in, const int* in_sizes, int n_in,
                              void* d_out, int out_size, void* d_ws, size_t ws_size,
                              hipStream_t stream) {
  const float* x     = (const float*)d_in[0];
  const float* w_q   = (const float*)d_in[1];
  const float* w_kv  = (const float*)d_in[2];
  const float* w_out = (const float*)d_in[3];
  float* out = (float*)d_out;

  __bf16* k_seq = (__bf16*)d_ws;                 // [2][4][16][256]
  __bf16* v_seq = k_seq + 2 * 4 * 16 * 256;      // [2][4][256][16]

  kv_conv_kernel<<<32, 256, 0, stream>>>(x, w_kv, k_seq, v_seq);
  attn_fused_kernel<<<16384, 32, 0, stream>>>(x, w_q, w_out, k_seq, v_seq, out);
}
